// EdgeModel_55499567399386
// MI455X (gfx1250) — compile-verified, matrix operand order checked by reading, not run on hardware
//
#include <hip/hip_runtime.h>
#include <hip/hip_bf16.h>

typedef __bf16 bf16_t;
typedef __attribute__((ext_vector_type(16))) __bf16 v16bf;
typedef __attribute__((ext_vector_type(8)))  __bf16 v8bf;
typedef __attribute__((ext_vector_type(8)))  float   v8f;

#define E_TOTAL 500000
#define IN_CH   512
#define OUTC    128
#define MT      64          // edge rows per block
#define XPAD    520         // padded LDS row stride (bf16 elems): rows land 4 banks apart

static __device__ __forceinline__ unsigned short f2bf_bits(float f) {
    unsigned u = __builtin_bit_cast(unsigned, f);
    unsigned r = (u + 0x7FFFu + ((u >> 16) & 1u)) >> 16;   // round-to-nearest-even
    return (unsigned short)r;
}
static __device__ __forceinline__ bf16_t f2bf(float f) {
    unsigned short s = f2bf_bits(f);
    return __builtin_bit_cast(bf16_t, s);
}

// 16-elem bf16 fragment from two 16-byte chunks (K 0..7 / K 16..23 per lane-half)
static __device__ __forceinline__ v16bf ldfrag(const bf16_t* p) {
    v8bf lo = *(const v8bf*)p;
    v8bf hi = *(const v8bf*)(p + 16);
    return __builtin_shufflevector(lo, hi, 0,1,2,3,4,5,6,7,8,9,10,11,12,13,14,15);
}

// ---------------- weight transpose + bf16 convert (runs once per launch) ----------------
__global__ __launch_bounds__(256) void prep_weights(
    const float* __restrict__ W1, const float* __restrict__ W2, const float* __restrict__ W3,
    bf16_t* __restrict__ w1t, bf16_t* __restrict__ w2t, bf16_t* __restrict__ w3t)
{
    int idx = blockIdx.x * 256 + threadIdx.x;
    if (idx < 512*512) {
        int n = idx >> 9, k = idx & 511;                 // w1t[n][k] = W1[k][n]
        w1t[idx] = f2bf(W1[k*512 + n]);
    } else if (idx < 2*512*512) {
        int i = idx - 512*512;
        int n = i >> 9, k = i & 511;
        w2t[i] = f2bf(W2[k*512 + n]);
    } else if (idx < 2*512*512 + 128*512) {
        int i = idx - 2*512*512;
        int n = i >> 9, k = i & 511;                     // n < 128
        w3t[i] = f2bf(W3[k*128 + n]);
    }
}

// One 512->512 layer: this wave owns 64 output cols (4 N-tiles, 2 groups of 2),
// all 4 M-subtiles of the 64-row block. B fragments double-buffered over k;
// k-loop unrolled x2 so the buffer swap is register renaming, not v_movs.
static __device__ __forceinline__ void layer512(
    const bf16_t* __restrict__ inL,    // LDS [MT][XPAD]
    bf16_t*       __restrict__ outL,   // LDS [MT][XPAD]
    const bf16_t* __restrict__ wt,     // [512][512] bf16, row = out col
    const float*  __restrict__ bias,
    int wave, int l15, int khalf)
{
    #pragma unroll
    for (int g = 0; g < 2; ++g) {
        const int nt0 = wave * 4 + g * 2;            // two 16-col N-tiles
        const bf16_t* bptr0 = wt + (long)(nt0*16      + l15) * IN_CH + khalf*8;
        const bf16_t* bptr1 = wt + (long)(nt0*16 + 16 + l15) * IN_CH + khalf*8;

        v8f acc[4][2];
        #pragma unroll
        for (int t = 0; t < 2; ++t) {
            float bv = bias[(nt0 + t)*16 + l15];
            #pragma unroll
            for (int m = 0; m < 4; ++m)
                #pragma unroll
                for (int j = 0; j < 8; ++j) acc[m][t][j] = bv;
        }

        v16bf b0 = ldfrag(bptr0);
        v16bf b1 = ldfrag(bptr1);

        #pragma unroll 2
        for (int kt = 0; kt < IN_CH/32; ++kt) {
            // A fragments for the 4 M-subtiles (LDS)
            v16bf a[4];
            #pragma unroll
            for (int m = 0; m < 4; ++m)
                a[m] = ldfrag(&inL[(m*16 + l15)*XPAD + kt*32 + khalf*8]);

            // prefetch next k-step's B fragments (wraps harmlessly on last step)
            const int kn = (kt + 1) & (IN_CH/32 - 1);
            v16bf nb0 = ldfrag(bptr0 + kn*32);
            v16bf nb1 = ldfrag(bptr1 + kn*32);

            #pragma unroll
            for (int m = 0; m < 4; ++m) {
                acc[m][0] = __builtin_amdgcn_wmma_f32_16x16x32_bf16(
                                false, a[m], false, b0, (short)0, acc[m][0], false, false);
                acc[m][1] = __builtin_amdgcn_wmma_f32_16x16x32_bf16(
                                false, a[m], false, b1, (short)0, acc[m][1], false, false);
            }
            b0 = nb0; b1 = nb1;
        }

        // relu + bf16 store to output LDS buffer
        #pragma unroll
        for (int t = 0; t < 2; ++t) {
            int ncol = (nt0 + t)*16 + l15;
            #pragma unroll
            for (int m = 0; m < 4; ++m)
                #pragma unroll
                for (int j = 0; j < 8; ++j) {
                    float v = acc[m][t][j]; v = v > 0.f ? v : 0.f;
                    outL[(m*16 + j + 8*khalf)*XPAD + ncol] = f2bf(v);
                }
        }
    }
}

// ---------------- fused 3-layer edge MLP ----------------
__global__ __launch_bounds__(256) void edge_mlp(
    const float* __restrict__ src, const float* __restrict__ dst,
    const float* __restrict__ ea,  const float* __restrict__ u,
    const int*   __restrict__ batch,
    const float* __restrict__ b1, const float* __restrict__ b2, const float* __restrict__ b3,
    const bf16_t* __restrict__ w1t, const bf16_t* __restrict__ w2t, const bf16_t* __restrict__ w3t,
    float* __restrict__ out)
{
    __shared__ bf16_t xs[MT * XPAD];   // ping
    __shared__ bf16_t hs[MT * XPAD];   // pong

    const int tid   = threadIdx.x;
    const int wave  = tid >> 5;
    const int lane  = tid & 31;
    const int l15   = lane & 15;
    const int khalf = lane >> 4;

    const long e0 = (long)blockIdx.x * MT;
    int rows = E_TOTAL - (int)e0; if (rows > MT) rows = MT;

    // ---- gather + concat + fp32->bf16 into LDS (float4 granularity) ----
    for (int idx = tid; idx < MT * (IN_CH/4); idx += 256) {
        int r = idx >> 7;             // IN_CH/4 == 128
        int c = (idx & 127) * 4;
        float4 v = make_float4(0.f, 0.f, 0.f, 0.f);
        if (r < rows) {
            long e = e0 + r;
            if      (c < 128) v = *(const float4*)(src + e*128 + c);
            else if (c < 256) v = *(const float4*)(dst + e*128 + (c-128));
            else if (c < 384) v = *(const float4*)(ea  + e*128 + (c-256));
            else { int bi = batch[e]; v = *(const float4*)(u + (long)bi*128 + (c-384)); }
        }
        ushort4 pk;
        pk.x = f2bf_bits(v.x); pk.y = f2bf_bits(v.y);
        pk.z = f2bf_bits(v.z); pk.w = f2bf_bits(v.w);
        *(ushort4*)&xs[r*XPAD + c] = pk;
    }
    __syncthreads();

    layer512(xs, hs, w1t, b1, wave, l15, khalf);   // layer 1
    __syncthreads();
    layer512(hs, xs, w2t, b2, wave, l15, khalf);   // layer 2
    __syncthreads();

    // ---------------- layer 3: xs(512) -> out(128 fp32) ----------------
    {
        const int nt = wave;                          // one 16-col N-tile per wave
        const bf16_t* bptr = w3t + (long)(nt*16 + l15) * 512 + khalf*8;

        v8f acc[4];
        {
            float bv = b3[nt*16 + l15];
            #pragma unroll
            for (int m = 0; m < 4; ++m)
                #pragma unroll
                for (int j = 0; j < 8; ++j) acc[m][j] = bv;
        }

        v16bf b = ldfrag(bptr);
        #pragma unroll 2
        for (int kt = 0; kt < 512/32; ++kt) {
            v16bf a[4];
            #pragma unroll
            for (int m = 0; m < 4; ++m)
                a[m] = ldfrag(&xs[(m*16 + l15)*XPAD + kt*32 + khalf*8]);

            const int kn = (kt + 1) & 15;
            v16bf nb = ldfrag(bptr + kn*32);

            #pragma unroll
            for (int m = 0; m < 4; ++m)
                acc[m] = __builtin_amdgcn_wmma_f32_16x16x32_bf16(
                             false, a[m], false, b, (short)0, acc[m], false, false);
            b = nb;
        }

        int ncol = nt*16 + l15;
        #pragma unroll
        for (int m = 0; m < 4; ++m)
            #pragma unroll
            for (int j = 0; j < 8; ++j) {
                int r = m*16 + j + 8*khalf;
                if (r < rows) out[(e0 + r)*(long)OUTC + ncol] = acc[m][j];
            }
    }
}

extern "C" void kernel_launch(void* const* d_in, const int* in_sizes, int n_in,
                              void* d_out, int out_size, void* d_ws, size_t ws_size,
                              hipStream_t stream) {
    const float* src   = (const float*)d_in[0];
    const float* dst   = (const float*)d_in[1];
    const float* ea    = (const float*)d_in[2];
    const float* u     = (const float*)d_in[3];
    const int*   batch = (const int*)  d_in[4];
    const float* W1    = (const float*)d_in[5];
    const float* b1    = (const float*)d_in[6];
    const float* W2    = (const float*)d_in[7];
    const float* b2    = (const float*)d_in[8];
    const float* W3    = (const float*)d_in[9];
    const float* b3    = (const float*)d_in[10];

    bf16_t* w1t = (bf16_t*)d_ws;                 // 512*512 bf16
    bf16_t* w2t = w1t + 512*512;                 // 512*512 bf16
    bf16_t* w3t = w2t + 512*512;                 // 128*512 bf16

    const int prep_elems = 2*512*512 + 128*512;  // 589824
    prep_weights<<<prep_elems/256, 256, 0, stream>>>(W1, W2, W3, w1t, w2t, w3t);

    const int blocks = (E_TOTAL + MT - 1) / MT;  // 7813
    edge_mlp<<<blocks, 256, 0, stream>>>(src, dst, ea, u, batch,
                                         b1, b2, b3, w1t, w2t, w3t,
                                         (float*)d_out);
}
